// ModulatedConv2d_31610959299063
// MI455X (gfx1250) — matrix operand order repaired and checked by reading, not compiled, not run on hardware
//
#include <hip/hip_runtime.h>

typedef __attribute__((ext_vector_type(16))) _Float16 v16h;
typedef __attribute__((ext_vector_type(8)))  float    v8f;
typedef __attribute__((ext_vector_type(4)))  int      v4i;

#define NB    16
#define CIN   512
#define COUT  512
#define HH    64
#define WW    64
#define SDIM  512
#define KTOT  4608              /* CIN * 9 */
#define HW    4096
#define PR    66                /* padded rows  (-1..64) */
#define PC    66                /* padded cols  (-1..64) */

#if __has_builtin(__builtin_amdgcn_global_load_async_to_lds_b128)
#define HAVE_ASYNC_LDS 1
#else
#define HAVE_ASYNC_LDS 0
#endif

#if __has_builtin(__builtin_amdgcn_s_wait_asynccnt)
#define WAIT_ASYNC(n) __builtin_amdgcn_s_wait_asynccnt(n)
#else
#define WAIT_ASYNC(n) asm volatile("s_wait_asynccnt %0" :: "i"(n) : "memory")
#endif

#if HAVE_ASYNC_LDS
typedef __attribute__((address_space(1))) v4i gas_v4i;   // global (prints __device__)
typedef __attribute__((address_space(3))) v4i las_v4i;   // LDS    (prints __shared__)
__device__ __forceinline__ void async_b128(const _Float16* g, _Float16* l) {
  __builtin_amdgcn_global_load_async_to_lds_b128(
      (gas_v4i*)g, (las_v4i*)l, 0, 0);
}
#endif

__device__ __forceinline__ float elr_scale() { return 0.044194173824159216f; } /* 1/sqrt(512) */
__device__ __forceinline__ float mod_scale() { return 6.5104166666666667e-4f; } /* 1/1536 */

// ---------------------------------------------------------------------------
// Kernel 1: s[b][i] = (style[b] . affine_w[i]) * ELR + affine_b[i] + 1
// ---------------------------------------------------------------------------
__global__ void style_affine_kernel(const float* __restrict__ style,
                                    const float* __restrict__ aw,
                                    const float* __restrict__ ab,
                                    float* __restrict__ s) {
  int gid = blockIdx.x * 256 + threadIdx.x;   // 8192 total
  int b = gid >> 9;
  int i = gid & 511;
  const float* st = style + (size_t)b * SDIM;
  const float* w  = aw + (size_t)i * SDIM;
  float acc = 0.f;
#pragma unroll 4
  for (int j = 0; j < SDIM; ++j) acc += st[j] * w[j];
  s[gid] = acc * elr_scale() + ab[i] + 1.0f;
}

// ---------------------------------------------------------------------------
// Kernel 2: zero-padded channel-last f16 activations:
//   xcl[b][row][col][cin], row/col in [0,66), interior = x[b][cin][row-1][col-1]
// ---------------------------------------------------------------------------
__global__ void xpad_cl_kernel(const float* __restrict__ x,
                               _Float16* __restrict__ xcl) {
  int pix = blockIdx.x;                 // row*66 + col
  int b   = blockIdx.y;
  int row = pix / PC;
  int col = pix - row * PC;
  int ih = row - 1, iw = col - 1;
  bool ok = ((unsigned)ih < (unsigned)HH) && ((unsigned)iw < (unsigned)WW);
  _Float16* o = xcl + (((size_t)b * PR + row) * PC + col) * CIN;
  const float* xb = x + (size_t)b * CIN * HW + ih * WW + iw;
  for (int c = threadIdx.x; c < CIN; c += 256) {
    float v = ok ? xb[(size_t)c * HW] : 0.f;
    o[c] = (_Float16)v;
  }
}

// ---------------------------------------------------------------------------
// Kernel 3: demod + f16 modulated weights in tap-major order:
//   wmod[b][cout][tap*512 + cin] = weight[cout][cin][tap] * s[b][cin] * MOD * d
// ---------------------------------------------------------------------------
__global__ void modulate_kernel(const float* __restrict__ weight,
                                const float* __restrict__ s,
                                _Float16* __restrict__ wmod) {
  __shared__ float red[256];
  int bc   = blockIdx.x;            // b*512 + cout
  int b    = bc >> 9;
  int cout = bc & 511;
  int t    = threadIdx.x;
  const float* wrow = weight + (size_t)cout * KTOT;   // [cin][9] layout
  const float* sb   = s + (size_t)b * CIN;

  float partial = 0.f;
  for (int idx = t; idx < KTOT; idx += 256) {
    int cin = idx / 9;
    float v = wrow[idx] * sb[cin];
    partial += v * v;
  }
  red[t] = partial;
  __syncthreads();
  for (int off = 128; off > 0; off >>= 1) {
    if (t < off) red[t] += red[t + off];
    __syncthreads();
  }
  float m = mod_scale();
  float d = rsqrtf(red[0] * m * m + 1e-4f) * m;

  _Float16* orow = wmod + (size_t)bc * KTOT;          // [tap][cin] layout
  for (int j = t; j < KTOT; j += 256) {
    int cin = j & 511;
    int tap = j >> 9;
    orow[j] = (_Float16)(wrow[cin * 9 + tap] * sb[cin] * d);
  }
}

// ---------------------------------------------------------------------------
// Kernel 4: implicit-GEMM grouped conv, v_wmma_f32_16x16x32_f16.
// Grid (h=64, mtile=4, b=16), block 256 = 8 wave32.
// Tile M=128 couts x N=64 pixels, K = 9 taps x 512 cin, step 32.
// Async global->LDS (double-buffered, ASYNCcnt) when available.
// ---------------------------------------------------------------------------
__global__ __launch_bounds__(256)
void conv_kernel(const _Float16* __restrict__ wmod,
                 const _Float16* __restrict__ xcl,
                 const float* __restrict__ bias,
                 float* __restrict__ out) {
#if HAVE_ASYNC_LDS
  __shared__ __align__(16) _Float16 ldsA[2][128 * 32];  // 16 KB
  __shared__ __align__(16) _Float16 ldsB[2][64 * 32];   //  8 KB
#else
  __shared__ __align__(16) _Float16 ldsA[1][128 * 32];
  __shared__ __align__(16) _Float16 ldsB[1][64 * 32];
#endif

  const int h     = blockIdx.x;
  const int mtile = blockIdx.y;
  const int b     = blockIdx.z;
  const int t     = threadIdx.x;
  const int lane  = t & 31;
  const int wv    = t >> 5;

  const int am = t >> 1;            // A-stage: weight row, 16 f16 at k-offset ak
  const int ak = (t & 1) << 4;
  const int bn  = t >> 2;           // B-stage: pixel, 8 cin at offset bko
  const int bko = (t & 3) << 3;

  const int mr  = lane & 15;
  const int kb8 = (lane >> 4) << 3;   // A-fragment K base (0/8)
  const int k16 = (lane >> 4) << 4;   // unused; B fragment uses kb8 too

  const _Float16* Arow = wmod + (size_t)(b * COUT + mtile * 128 + am) * KTOT + ak;
  const _Float16* Xb   = xcl + (size_t)b * PR * PC * CIN;

  _Float16* dstA = &ldsA[0][am * 32 + ak];
  _Float16* dstB = &ldsB[0][bn * 32 + bko];
  const int bufA = 128 * 32;        // element stride between LDS buffers
  const int bufB = 64 * 32;

  v8f acc[4] = {};

#if HAVE_ASYNC_LDS
  int p = 0;
  for (int tap = 0; tap < 9; ++tap) {
    const int kh  = tap / 3;
    const int kw  = tap - kh * 3;
    const _Float16* Ab = Arow + tap * CIN;
    const _Float16* Bb = Xb + (((size_t)(h + kh)) * PC + (bn + kw)) * CIN + bko;

    // prologue: issue c0 = 0 into buffer p (safe: all waves finished reading
    // buffer p two steps ago, enforced by the per-step barrier chain)
    async_b128(Ab,     dstA + p * bufA);
    async_b128(Ab + 8, dstA + p * bufA + 8);
    async_b128(Bb,     dstB + p * bufB);

    for (int c0 = 0; c0 < CIN; c0 += 32) {
      WAIT_ASYNC(0);          // my loads for buffer p have landed in LDS
      __syncthreads();        // everyone's loads landed + prev compute retired

      if (c0 + 32 < CIN) {    // issue next step into the other buffer
        const int q = p ^ 1;
        async_b128(Ab + c0 + 32,     dstA + q * bufA);
        async_b128(Ab + c0 + 32 + 8, dstA + q * bufA + 8);
        async_b128(Bb + c0 + 32,     dstB + q * bufB);
      }

      // A fragment: lane holds K = kb8..kb8+7 and 16+kb8..+7 of row mr
      union { v16h v; uint4 q[2]; } af;
      const _Float16* ar = &ldsA[p][(wv * 16 + mr) * 32];
      af.q[0] = *(const uint4*)(ar + kb8);
      af.q[1] = *(const uint4*)(ar + kb8 + 16);

#pragma unroll
      for (int nt = 0; nt < 4; ++nt) {
        union { v16h v; uint4 q[2]; } bf;
        const _Float16* br = &ldsB[p][(nt * 16 + mr) * 32 + kb8];
        bf.q[0] = *(const uint4*)(br);
        bf.q[1] = *(const uint4*)(br + 16);
        acc[nt] = __builtin_amdgcn_wmma_f32_16x16x32_f16(
            false, af.v, false, bf.v, (short)0, acc[nt], false, false);
      }
      p ^= 1;
    }
  }
#else
  // synchronous fallback
  for (int tap = 0; tap < 9; ++tap) {
    const int kh  = tap / 3;
    const int kw  = tap - kh * 3;
    const _Float16* Ab = Arow + tap * CIN;
    const _Float16* Bb = Xb + (((size_t)(h + kh)) * PC + (bn + kw)) * CIN + bko;

    for (int c0 = 0; c0 < CIN; c0 += 32) {
      __syncthreads();
      {
        const uint4* ga = (const uint4*)(Ab + c0);
        uint4* da = (uint4*)dstA;
        da[0] = ga[0];
        da[1] = ga[1];
      }
      *(uint4*)dstB = *(const uint4*)(Bb + c0);
      __syncthreads();

      union { v16h v; uint4 q[2]; } af;
      const _Float16* ar = &ldsA[0][(wv * 16 + mr) * 32];
      af.q[0] = *(const uint4*)(ar + kb8);
      af.q[1] = *(const uint4*)(ar + kb8 + 16);

#pragma unroll
      for (int nt = 0; nt < 4; ++nt) {
        union { v16h v; uint4 q[2]; } bf;
        const _Float16* br = &ldsB[0][(nt * 16 + mr) * 32 + kb8];
        bf.q[0] = *(const uint4*)(br);
        bf.q[1] = *(const uint4*)(br + 16);
        acc[nt] = __builtin_amdgcn_wmma_f32_16x16x32_f16(
            false, af.v, false, bf.v, (short)0, acc[nt], false, false);
      }
    }
  }
#endif

  // store: C/D layout N = lane&15, M = r8 + 8*(lane>=16)
  const int mbase = mtile * 128 + wv * 16 + ((lane >> 4) << 3);
  float* ob = out + (((size_t)b * COUT + mbase) * HH + h) * WW + (lane & 15);
  const float* bb = bias + mbase;
#pragma unroll
  for (int nt = 0; nt < 4; ++nt) {
#pragma unroll
    for (int r8 = 0; r8 < 8; ++r8) {
      ob[(size_t)r8 * HW + (nt << 4)] = acc[nt][r8] + bb[r8];
    }
  }
}

// ---------------------------------------------------------------------------
extern "C" void kernel_launch(void* const* d_in, const int* in_sizes, int n_in,
                              void* d_out, int out_size, void* d_ws, size_t ws_size,
                              hipStream_t stream) {
  const float* x        = (const float*)d_in[0];
  const float* style    = (const float*)d_in[1];
  const float* affine_w = (const float*)d_in[2];
  const float* affine_b = (const float*)d_in[3];
  const float* weight   = (const float*)d_in[4];
  const float* bias     = (const float*)d_in[5];
  float* out = (float*)d_out;

  // workspace carve-up (~143.4 MB)
  char* ws = (char*)d_ws;
  float*    s    = (float*)ws;                                  // 32 KB
  _Float16* wmod = (_Float16*)(ws + 32 * 1024);                 // 72 MB
  _Float16* xcl  = (_Float16*)(ws + 32 * 1024
                               + (size_t)NB * COUT * KTOT * 2); // 68 MB

  style_affine_kernel<<<dim3(32), dim3(256), 0, stream>>>(style, affine_w, affine_b, s);

  xpad_cl_kernel<<<dim3(PR * PC, NB), dim3(256), 0, stream>>>(x, xcl);

  modulate_kernel<<<dim3(NB * COUT), dim3(256), 0, stream>>>(weight, s, wmod);

  conv_kernel<<<dim3(HH, COUT / 128, NB), dim3(256), 0, stream>>>(wmod, xcl, bias, out);

  (void)in_sizes; (void)n_in; (void)out_size; (void)ws_size;
}
